// ConvNextBlock_64905545777173
// MI455X (gfx1250) — matrix operand order repaired and checked by reading, not compiled, not run on hardware
//
#include <hip/hip_runtime.h>
#include <math.h>

// ---------------------------------------------------------------------------
// ConvNeXt block for MI455X (gfx1250, wave32, WMMA).
//   x:      (2, 256, 256, 256) f32
//   stage1: depthwise 7x7 conv + bias + LayerNorm(W axis)  -> A (bf16, 131072x256)
//   stage2: per-row MLP 256 -> 1024 -GELU-> 256 via v_wmma_f32_16x16x32_bf16,
//           fused (hidden tensor never hits HBM), + transpose + residual.
// ---------------------------------------------------------------------------

typedef __attribute__((ext_vector_type(16))) __bf16 v16bf;
typedef __attribute__((ext_vector_type(8)))  __bf16 v8bf;
typedef __attribute__((ext_vector_type(8)))  float  v8f;

#define DIMX 256
#define NROWS (2 * 256 * 256)   /* 131072 GEMM rows */
#define HID   1024

// Build a WMMA 16-element fragment from two 16-byte loads (A and B frags
// both take 16 contiguous K values per lane half, k = 16*half + s).
__device__ __forceinline__ v16bf frag_2x8(const __bf16* p0, const __bf16* p1) {
  v8bf lo = *(const v8bf*)p0;
  v8bf hi = *(const v8bf*)p1;
  v16bf r;
#pragma unroll
  for (int i = 0; i < 8; ++i) { r[i] = lo[i]; r[i + 8] = hi[i]; }
  return r;
}

__device__ __forceinline__ float gelu_exact(float t) {
  return 0.5f * t * (1.0f + erff(t * 0.70710678118654752f));
}

// ---------------------------------------------------------------------------
// Kernel 1: depthwise 7x7 conv + bias + LayerNorm over W, emit bf16 row.
// One block per (b, c, h) row; 256 threads (one per w).
// ---------------------------------------------------------------------------
__global__ void __launch_bounds__(256)
conv_ln_kernel(const float* __restrict__ x, const float* __restrict__ conv_w,
               const float* __restrict__ conv_b, const float* __restrict__ ln_g,
               const float* __restrict__ ln_b, __bf16* __restrict__ A) {
  __shared__ float rows[7][DIMX];
  __shared__ float wk[49];
  __shared__ float red[DIMX];
  __shared__ float stats[2];

  const int t   = threadIdx.x;
  const int blk = blockIdx.x;
  const int h   = blk & 255;
  const int c   = (blk >> 8) & 255;
  const int b   = blk >> 16;
  const size_t planeBase = ((size_t)(b * DIMX + c)) * DIMX * DIMX;

#pragma unroll
  for (int dy = 0; dy < 7; ++dy) {
    const int hin = h + dy - 3;
    rows[dy][t] = (hin >= 0 && hin < DIMX)
                    ? x[planeBase + (size_t)hin * DIMX + t] : 0.0f;
  }
  if (t < 49) wk[t] = conv_w[c * 49 + t];
  __syncthreads();

  float acc = 0.0f;
#pragma unroll
  for (int dy = 0; dy < 7; ++dy) {
#pragma unroll
    for (int dx = 0; dx < 7; ++dx) {
      const int wc = t + dx - 3;
      if (wc >= 0 && wc < DIMX) acc += rows[dy][wc] * wk[dy * 7 + dx];
    }
  }
  acc += conv_b[c];

  // mean over W
  red[t] = acc;
  __syncthreads();
  for (int s = 128; s > 0; s >>= 1) {
    if (t < s) red[t] += red[t + s];
    __syncthreads();
  }
  if (t == 0) stats[0] = red[0] * (1.0f / DIMX);
  __syncthreads();
  const float mu = stats[0];
  const float d  = acc - mu;

  // variance over W
  red[t] = d * d;
  __syncthreads();
  for (int s = 128; s > 0; s >>= 1) {
    if (t < s) red[t] += red[t + s];
    __syncthreads();
  }
  if (t == 0) stats[1] = red[0] * (1.0f / DIMX);
  __syncthreads();

  const float inv = rsqrtf(stats[1] + 1e-5f);
  const float val = d * inv * ln_g[t] + ln_b[t];
  A[(size_t)blk * DIMX + t] = (__bf16)val;
}

// ---------------------------------------------------------------------------
// Kernel 2: f32 -> bf16 weight conversion (w1: 1024x256, w2: 256x1024;
// both K-contiguous, which is exactly the WMMA B-fragment friendly layout).
// ---------------------------------------------------------------------------
__global__ void __launch_bounds__(256)
cvt_weights_kernel(const float* __restrict__ w1, const float* __restrict__ w2,
                   __bf16* __restrict__ W1, __bf16* __restrict__ W2) {
  const int i = blockIdx.x * 256 + threadIdx.x;  // grid covers 262144 exactly
  W1[i] = (__bf16)w1[i];
  W2[i] = (__bf16)w2[i];
}

// ---------------------------------------------------------------------------
// Kernel 3: fused MLP. Block = 8 waves = 256 threads, owns 32 rows x 256 outs.
// Hidden dim processed in 8 chunks of 128:
//   phase A: all waves compute 32x128 GELU(A @ W1^T + b1) chunk -> LDS (bf16)
//   phase B: wave (mi,ni) accumulates O[16mi:+16, 64ni:+64] += T @ W2chunk
// Epilogue: LDS transpose -> coalesced out[b, n, cY, hY] = x + O + b2.
// ---------------------------------------------------------------------------
__global__ void __launch_bounds__(256)
mlp_wmma_kernel(const __bf16* __restrict__ A, const __bf16* __restrict__ W1,
                const __bf16* __restrict__ W2, const float* __restrict__ b1,
                const float* __restrict__ b2, const float* __restrict__ x,
                float* __restrict__ out) {
  __shared__ __bf16 Ash[32][264];   // A tile (32x256, +8 pad)
  __shared__ __bf16 Tsh[32][136];   // GELU'd hidden chunk (32x128, +8 pad)
  __shared__ float  Osh[32][257];   // output tile for transpose (+1 pad)

  const int tid  = threadIdx.x;
  const int wv   = tid >> 5;
  const int lane = tid & 31;
  const int l16  = lane & 15;
  const int half = lane >> 4;
  const int mi   = wv >> 2;   // 0..1 : row strip
  const int ni   = wv & 3;    // 0..3 : col strip

  const int rBase = blockIdx.x * 32;
  const int bIdx  = rBase >> 16;
  const int cY    = (rBase >> 8) & 255;
  const int hY0   = rBase & 255;

  // Stage the 32x256 bf16 A tile into LDS (uint4 granularity, read-once from HBM).
  {
    const uint4* src = (const uint4*)(A + (size_t)rBase * 256);
#pragma unroll
    for (int idx = tid; idx < 1024; idx += 256) {
      const int r = idx >> 5, c4 = idx & 31;
      ((uint4*)&Ash[r][0])[c4] = src[r * 32 + c4];
    }
  }
  __syncthreads();

  v8f oc[4];
#pragma unroll
  for (int s = 0; s < 4; ++s)
#pragma unroll
    for (int j = 0; j < 8; ++j) oc[s][j] = 0.0f;

  const int aRow = 16 * mi + l16;

  for (int chunk = 0; chunk < 8; ++chunk) {
    const int oBase = chunk * 128;

    // ---- phase A: this wave computes T rows 16mi:+16, cols 32ni:+32 ----
    v8f c0, c1;
#pragma unroll
    for (int j = 0; j < 8; ++j) { c0[j] = 0.0f; c1[j] = 0.0f; }

#pragma unroll
    for (int k1 = 0; k1 < 8; ++k1) {
      const int k0 = k1 * 32 + 16 * half;             // A frag: 16/16 K split
      const v16bf af = frag_2x8(&Ash[aRow][k0], &Ash[aRow][k0 + 8]);
      const int kB = k1 * 32 + 16 * half;             // B frag: 16/16 K split
      const __bf16* bp0 = W1 + (size_t)(oBase + 32 * ni + l16) * 256 + kB;
      const __bf16* bp1 = W1 + (size_t)(oBase + 32 * ni + 16 + l16) * 256 + kB;
      const v16bf bf0 = frag_2x8(bp0, bp0 + 8);
      const v16bf bf1 = frag_2x8(bp1, bp1 + 8);
      c0 = __builtin_amdgcn_wmma_f32_16x16x32_bf16(false, af, false, bf0,
                                                   (short)0, c0, false, false);
      c1 = __builtin_amdgcn_wmma_f32_16x16x32_bf16(false, af, false, bf1,
                                                   (short)0, c1, false, false);
    }

    __syncthreads();  // WAR: previous phase B finished reading Tsh

    {
      const int n0 = oBase + 32 * ni + l16;
      const float bias0 = b1[n0], bias1 = b1[n0 + 16];
#pragma unroll
      for (int v = 0; v < 8; ++v) {
        const int tr = 16 * mi + v + 8 * half;        // C frag row mapping
        Tsh[tr][32 * ni + l16]      = (__bf16)gelu_exact(c0[v] + bias0);
        Tsh[tr][32 * ni + 16 + l16] = (__bf16)gelu_exact(c1[v] + bias1);
      }
    }
    __syncthreads();  // RAW: Tsh chunk complete

    // ---- phase B: O[16mi:+16, 64ni:+64] += T(32x128) @ W2chunk(128x64) ----
#pragma unroll
    for (int k2 = 0; k2 < 4; ++k2) {
      const int k0 = k2 * 32 + 16 * half;
      const v16bf a2 = frag_2x8(&Tsh[aRow][k0], &Tsh[aRow][k0 + 8]);
      const int kB = oBase + k2 * 32 + 16 * half;
#pragma unroll
      for (int s = 0; s < 4; ++s) {
        const __bf16* bp = W2 + (size_t)(64 * ni + 16 * s + l16) * 1024 + kB;
        const v16bf bfr = frag_2x8(bp, bp + 8);
        oc[s] = __builtin_amdgcn_wmma_f32_16x16x32_bf16(false, a2, false, bfr,
                                                        (short)0, oc[s], false, false);
      }
    }
  }

  // ---- epilogue: LDS transpose, then coalesced residual writeback ----
#pragma unroll
  for (int s = 0; s < 4; ++s) {
    const int col = 64 * ni + 16 * s + l16;
#pragma unroll
    for (int v = 0; v < 8; ++v)
      Osh[16 * mi + v + 8 * half][col] = oc[s][v];
  }
  __syncthreads();

  // out[b, n, cY, hY0+lane] = x[...] + O[hY local = lane][n] + b2[n]
#pragma unroll 4
  for (int it = 0; it < 32; ++it) {
    const int n = it * 8 + wv;
    const float val = Osh[lane][n] + b2[n];
    const size_t oidx =
        (((size_t)bIdx * DIMX + n) * DIMX + cY) * DIMX + (size_t)(hY0 + lane);
    out[oidx] = x[oidx] + val;
  }
}

// ---------------------------------------------------------------------------
// Host launcher. Workspace layout:
//   [0, 64 MiB)      : A   (131072 x 256 bf16)
//   [+0.5 MiB)       : W1  (1024 x 256 bf16)
//   [+0.5 MiB)       : W2  (256 x 1024 bf16)
// ---------------------------------------------------------------------------
extern "C" void kernel_launch(void* const* d_in, const int* in_sizes, int n_in,
                              void* d_out, int out_size, void* d_ws,
                              size_t ws_size, hipStream_t stream) {
  const float* x      = (const float*)d_in[0];
  const float* conv_w = (const float*)d_in[1];
  const float* conv_b = (const float*)d_in[2];
  const float* ln_g   = (const float*)d_in[3];
  const float* ln_b   = (const float*)d_in[4];
  const float* w1     = (const float*)d_in[5];
  const float* b1     = (const float*)d_in[6];
  const float* w2     = (const float*)d_in[7];
  const float* b2     = (const float*)d_in[8];
  float* out = (float*)d_out;

  char* ws = (char*)d_ws;
  __bf16* A  = (__bf16*)ws;                                  // 67,108,864 B
  __bf16* W1 = (__bf16*)(ws + (size_t)67108864);             //    524,288 B
  __bf16* W2 = (__bf16*)(ws + (size_t)67108864 + 524288);    //    524,288 B

  hipLaunchKernelGGL(conv_ln_kernel, dim3(NROWS), dim3(256), 0, stream,
                     x, conv_w, conv_b, ln_g, ln_b, A);
  hipLaunchKernelGGL(cvt_weights_kernel, dim3(1024), dim3(256), 0, stream,
                     w1, w2, W1, W2);
  hipLaunchKernelGGL(mlp_wmma_kernel, dim3(NROWS / 32), dim3(256), 0, stream,
                     A, W1, W2, b1, b2, x, out);
}
